// AttentionModel_40312563041042
// MI455X (gfx1250) — matrix operand order repaired and checked by reading, not compiled, not run on hardware
//
#include <hip/hip_runtime.h>
#include <hip/hip_bf16.h>
#include <math.h>

#define S_STEPS 512
#define BATCH   256
#define N_IN    64
#define N_NEUR  256

typedef __bf16 bf16;
typedef __attribute__((ext_vector_type(16))) __bf16 v16bf;
typedef __attribute__((ext_vector_type(8)))  __bf16 v8bf;
typedef __attribute__((ext_vector_type(8)))  float  v8f;

union V16 { v16bf v; v8bf h[2]; };

// ---- WMMA helpers (CDNA5 16x16x32 bf16, f32 accumulate) -------------------
__device__ inline v8f wmma_bf16(v16bf a, v16bf b, v8f c) {
    // 8 args: (neg_a, A, neg_b, B, c_mod, C, reuse_a, reuse_b)
    return __builtin_amdgcn_wmma_f32_16x16x32_bf16(false, a, false, b,
                                                   (short)0, c, false, false);
}

// A fragment (16x32, MxK) from row-major bf16 src with leading dim ld.
// ISA layout: lanes 0-15 row M=lane, K={kb+half*8+0..7, kb+half*8+16..23}
__device__ inline v16bf load_A_rowmajor(const bf16* src, int ld, int kb, int lane) {
    int row = lane & 15, half = lane >> 4;
    const bf16* p = src + (size_t)row * ld + kb + half * 8;
    V16 u;
    u.h[0] = *(const v8bf*)(p);
    u.h[1] = *(const v8bf*)(p + 16);
    return u.v;
}

// A fragment converted on the fly from f32 source.
__device__ inline v16bf load_A_f32(const float* src, int ld, int kb, int lane) {
    int row = lane & 15, half = lane >> 4;
    const float* p = src + (size_t)row * ld + kb + half * 8;
    v16bf a;
#pragma unroll
    for (int e = 0; e < 8; ++e) a[e] = (bf16)p[e];
#pragma unroll
    for (int e = 0; e < 8; ++e) a[e + 8] = (bf16)p[e + 16];
    return a;
}

// B fragment (32x16, KxN) where B[k][n] = W[n][k] (W row-major NxK, ld=ldw).
// ISA layout: lane%16 = col N, K = {kb+half*8+0..7, kb+half*8+16..23}.
__device__ inline v16bf load_B_WT(const bf16* W, int ldw, int n0, int kb, int lane) {
    int col = lane & 15, half = lane >> 4;
    const bf16* p = W + (size_t)(n0 + col) * ldw + kb + half * 8;
    V16 u;
    u.h[0] = *(const v8bf*)(p);
    u.h[1] = *(const v8bf*)(p + 16);
    return u.v;
}

// ---- small f32 -> bf16 weight converter -----------------------------------
__global__ void k_cvt(const float* __restrict__ src, bf16* __restrict__ dst, int n) {
    int i = blockIdx.x * blockDim.x + threadIdx.x;
    if (i < n) dst[i] = (bf16)src[i];
}

// ---- Kernel 1: input projection XW[s*B+b][n] = X @ W_ih^T (parallel GEMM) --
__global__ void __launch_bounds__(256) k_proj(const float* __restrict__ X,
                                              const bf16* __restrict__ Wih,
                                              bf16* __restrict__ XW) {
    int lane = threadIdx.x & 31, w = threadIdx.x >> 5;
    int m0 = blockIdx.x * 16;            // flat (s*B+b) tile
    int col = lane & 15, half = lane >> 4;
    v16bf a0 = load_A_f32(X + (size_t)m0 * N_IN, N_IN, 0, lane);
    v16bf a1 = load_A_f32(X + (size_t)m0 * N_IN, N_IN, 32, lane);
#pragma unroll
    for (int i = 0; i < 2; ++i) {
        int n0 = (w * 2 + i) * 16;
        v8f c = {};
        c = wmma_bf16(a0, load_B_WT(Wih, N_IN, n0, 0, lane), c);
        c = wmma_bf16(a1, load_B_WT(Wih, N_IN, n0, 32, lane), c);
        int n = n0 + col;
#pragma unroll
        for (int r = 0; r < 8; ++r)
            XW[(size_t)(m0 + r + half * 8) * N_NEUR + n] = (bf16)c[r];
    }
}

// ---- Kernel 2: recurrent tanh RNN. 16 WGs x 16 batch rows, H in LDS. ------
__global__ void __launch_bounds__(256) k_rnn(const bf16* __restrict__ XW,
                                             const bf16* __restrict__ Whh,
                                             const float* __restrict__ b_ih,
                                             const float* __restrict__ b_hh,
                                             bf16* __restrict__ out_pre) {
    __shared__ __align__(16) bf16 Hsh[16 * N_NEUR];   // 8 KB
    int tid = threadIdx.x, lane = tid & 31, w = tid >> 5;
    int b0 = blockIdx.x * 16;
    int col = lane & 15, half = lane >> 4;

    for (int i = tid; i < 16 * N_NEUR; i += blockDim.x) Hsh[i] = (bf16)0.f;
    __syncthreads();

    int n0a = (w * 2) * 16, n0b = (w * 2 + 1) * 16;
    float bias0 = b_ih[n0a + col] + b_hh[n0a + col];
    float bias1 = b_ih[n0b + col] + b_hh[n0b + col];

    for (int s = 0; s < S_STEPS; ++s) {
        // read current H as A fragments (all K chunks) from LDS
        v16bf a[8];
#pragma unroll
        for (int kk = 0; kk < 8; ++kk)
            a[kk] = load_A_rowmajor(Hsh, N_NEUR, kk * 32, lane);
        __syncthreads();   // all waves done reading H before it is overwritten

        const bf16* xwrow = XW + ((size_t)s * BATCH + b0) * N_NEUR;
        float hv[2][8];
#pragma unroll
        for (int i = 0; i < 2; ++i) {
            int n0 = (w * 2 + i) * 16;
            int n = n0 + col;
            v8f c = {};
#pragma unroll
            for (int r = 0; r < 8; ++r)
                c[r] = (float)xwrow[(size_t)(r + half * 8) * N_NEUR + n] + (i ? bias1 : bias0);
#pragma unroll
            for (int kk = 0; kk < 8; ++kk)
                c = wmma_bf16(a[kk], load_B_WT(Whh, N_NEUR, n0, kk * 32, lane), c);
#pragma unroll
            for (int r = 0; r < 8; ++r) hv[i][r] = tanhf(c[r]);
        }
        bf16* orow = out_pre + ((size_t)s * BATCH + b0) * N_NEUR;
#pragma unroll
        for (int i = 0; i < 2; ++i) {
            int n = (w * 2 + i) * 16 + col;
#pragma unroll
            for (int r = 0; r < 8; ++r) {
                bf16 hb = (bf16)hv[i][r];
                Hsh[(r + half * 8) * N_NEUR + n] = hb;
                orow[(size_t)(r + half * 8) * N_NEUR + n] = hb;
            }
        }
        __syncthreads();
    }
}

// ---- Kernel 3: attention loop (256 sequential iters), per-WG batch slice. --
__global__ void __launch_bounds__(256) k_attn(const bf16* __restrict__ O,
                                              const bf16* __restrict__ Wcih,
                                              const bf16* __restrict__ Wchh,
                                              const float* __restrict__ bc_ih,
                                              const float* __restrict__ bc_hh,
                                              const float* __restrict__ W_fc,
                                              const float* __restrict__ b_fc,
                                              float* __restrict__ out) {
    __shared__ __align__(16) bf16 hp_sh[16 * N_NEUR];    // 8 KB
    __shared__ __align__(16) bf16 att_sh[16 * N_NEUR];   // 8 KB
    __shared__ float soft_sh[S_STEPS * 16];              // 32 KB
    int tid = threadIdx.x, lane = tid & 31, w = tid >> 5;
    int b0 = blockIdx.x * 16;
    int col = lane & 15, half = lane >> 4;

    for (int i = tid; i < 16 * N_NEUR; i += blockDim.x) hp_sh[i] = (bf16)0.f;
    __syncthreads();

    int n0a = (w * 2) * 16, n0b = (w * 2 + 1) * 16;
    float bias0 = bc_ih[n0a + col] + bc_hh[n0a + col];
    float bias1 = bc_ih[n0b + col] + bc_hh[n0b + col];

    for (int it = 0; it < N_NEUR; ++it) {
        // --- phase A: pre_soft[s][j] = sum_n O[s][b0+j][n] * hp[j][n]
        for (int i = 0; i < 32; ++i) {
            int p = i * 256 + tid;
            int s = p >> 4, j = p & 15;
            const bf16* orow = O + ((size_t)s * BATCH + b0 + j) * N_NEUR;
            const bf16* hrow = hp_sh + j * N_NEUR;
            float acc = 0.f;
            for (int n = 0; n < N_NEUR; n += 8) {
                v8bf ov = *(const v8bf*)(orow + n);
                v8bf hv = *(const v8bf*)(hrow + n);
#pragma unroll
                for (int e = 0; e < 8; ++e) acc += (float)ov[e] * (float)hv[e];
            }
            soft_sh[s * 16 + j] = acc;
        }
        __syncthreads();

        // --- phase B: softmax over s (per column j); 16 threads per column
        {
            int j = tid >> 4, lg = tid & 15;
            float mx = -1e30f;
            for (int k = 0; k < 32; ++k)
                mx = fmaxf(mx, soft_sh[(lg + 16 * k) * 16 + j]);
#pragma unroll
            for (int m = 1; m < 16; m <<= 1) mx = fmaxf(mx, __shfl_xor(mx, m, 16));
            float sum = 0.f;
            for (int k = 0; k < 32; ++k) {
                float e = __expf(soft_sh[(lg + 16 * k) * 16 + j] - mx);
                soft_sh[(lg + 16 * k) * 16 + j] = e;
                sum += e;
            }
#pragma unroll
            for (int m = 1; m < 16; m <<= 1) sum += __shfl_xor(sum, m, 16);
            float inv = 1.f / sum;
            for (int k = 0; k < 32; ++k) soft_sh[(lg + 16 * k) * 16 + j] *= inv;
        }
        __syncthreads();

        // --- phase C: attention[j][n] = sum_s O[s][b0+j][n] * soft[s][j]
        {
            int j = tid >> 4, nb = (tid & 15) * 16;
            float acc[16];
#pragma unroll
            for (int e = 0; e < 16; ++e) acc[e] = 0.f;
            for (int s = 0; s < S_STEPS; ++s) {
                float ws = soft_sh[s * 16 + j];
                const bf16* orow = O + ((size_t)s * BATCH + b0 + j) * N_NEUR + nb;
                if (s + 8 < S_STEPS)
                    __builtin_prefetch(orow + (size_t)8 * BATCH * N_NEUR, 0, 1);
                v8bf o0 = *(const v8bf*)(orow);
                v8bf o1 = *(const v8bf*)(orow + 8);
#pragma unroll
                for (int e = 0; e < 8; ++e) {
                    acc[e]     += ws * (float)o0[e];
                    acc[e + 8] += ws * (float)o1[e];
                }
            }
            bf16* arow = att_sh + j * N_NEUR + nb;
#pragma unroll
            for (int e = 0; e < 16; ++e) arow[e] = (bf16)acc[e];
        }
        __syncthreads();

        // --- phase D: hp = tanh(hp@Wcih^T + att@Wchh^T + biases)  (WMMA)
        float hv[2][8];
#pragma unroll
        for (int i = 0; i < 2; ++i) {
            int n0 = (w * 2 + i) * 16;
            v8f c = {};
#pragma unroll
            for (int r = 0; r < 8; ++r) c[r] = (i ? bias1 : bias0);
#pragma unroll
            for (int kk = 0; kk < 8; ++kk)
                c = wmma_bf16(load_A_rowmajor(hp_sh, N_NEUR, kk * 32, lane),
                              load_B_WT(Wcih, N_NEUR, n0, kk * 32, lane), c);
#pragma unroll
            for (int kk = 0; kk < 8; ++kk)
                c = wmma_bf16(load_A_rowmajor(att_sh, N_NEUR, kk * 32, lane),
                              load_B_WT(Wchh, N_NEUR, n0, kk * 32, lane), c);
#pragma unroll
            for (int r = 0; r < 8; ++r) hv[i][r] = tanhf(c[r]);
        }
        __syncthreads();   // all reads of hp_sh complete
#pragma unroll
        for (int i = 0; i < 2; ++i) {
            int n = (w * 2 + i) * 16 + col;
#pragma unroll
            for (int r = 0; r < 8; ++r)
                hp_sh[(r + half * 8) * N_NEUR + n] = (bf16)hv[i][r];
        }
        __syncthreads();
    }

    // --- final fc: out[b] = hp[b] . W_fc + b_fc
    if (tid < 16) {
        int j = tid;
        float acc = 0.f;
        for (int n = 0; n < N_NEUR; ++n)
            acc += (float)hp_sh[j * N_NEUR + n] * W_fc[n];
        out[b0 + j] = acc + b_fc[0];
    }
}

extern "C" void kernel_launch(void* const* d_in, const int* in_sizes, int n_in,
                              void* d_out, int out_size, void* d_ws, size_t ws_size,
                              hipStream_t stream) {
    const float* X    = (const float*)d_in[0];
    const float* Wih  = (const float*)d_in[1];
    const float* Whh  = (const float*)d_in[2];
    const float* bih  = (const float*)d_in[3];
    const float* bhh  = (const float*)d_in[4];
    const float* Wcih = (const float*)d_in[5];
    const float* Wchh = (const float*)d_in[6];
    const float* bcih = (const float*)d_in[7];
    const float* bchh = (const float*)d_in[8];
    const float* Wfc  = (const float*)d_in[9];
    const float* bfc  = (const float*)d_in[10];

    char* ws = (char*)d_ws;
    size_t off = 0;
    bf16* XW      = (bf16*)(ws + off); off += (size_t)S_STEPS * BATCH * N_NEUR * 2;
    bf16* O       = (bf16*)(ws + off); off += (size_t)S_STEPS * BATCH * N_NEUR * 2;
    bf16* Wih_bf  = (bf16*)(ws + off); off += (size_t)N_NEUR * N_IN * 2;
    bf16* Whh_bf  = (bf16*)(ws + off); off += (size_t)N_NEUR * N_NEUR * 2;
    bf16* Wcih_bf = (bf16*)(ws + off); off += (size_t)N_NEUR * N_NEUR * 2;
    bf16* Wchh_bf = (bf16*)(ws + off); off += (size_t)N_NEUR * N_NEUR * 2;

    k_cvt<<<(N_NEUR * N_IN + 255) / 256, 256, 0, stream>>>(Wih, Wih_bf, N_NEUR * N_IN);
    k_cvt<<<(N_NEUR * N_NEUR + 255) / 256, 256, 0, stream>>>(Whh, Whh_bf, N_NEUR * N_NEUR);
    k_cvt<<<(N_NEUR * N_NEUR + 255) / 256, 256, 0, stream>>>(Wcih, Wcih_bf, N_NEUR * N_NEUR);
    k_cvt<<<(N_NEUR * N_NEUR + 255) / 256, 256, 0, stream>>>(Wchh, Wchh_bf, N_NEUR * N_NEUR);

    k_proj<<<(S_STEPS * BATCH) / 16, 256, 0, stream>>>(X, Wih_bf, XW);
    k_rnn<<<BATCH / 16, 256, 0, stream>>>(XW, Whh_bf, bih, bhh, O);
    k_attn<<<BATCH / 16, 256, 0, stream>>>(O, Wcih_bf, Wchh_bf, bcih, bchh,
                                           Wfc, bfc, (float*)d_out);
}
